// PatchConv2Layer_8117488190080
// MI455X (gfx1250) — matrix operand order repaired and checked by reading, not compiled, not run on hardware
//
#include <hip/hip_runtime.h>

typedef __attribute__((ext_vector_type(16))) _Float16 v16h;
typedef __attribute__((ext_vector_type(8)))  float    v8f;

#define NEG_SLOPE 0.01f

__device__ __forceinline__ float lrelu(float v) { return v > 0.0f ? v : NEG_SLOPE * v; }

// ---------------- utility kernels ----------------

__global__ void k_zero(float* __restrict__ p, long long n) {
  long long i = (long long)blockIdx.x * blockDim.x + threadIdx.x;
  long long s = (long long)gridDim.x * blockDim.x;
  for (; i < n; i += s) p[i] = 0.0f;
}

__global__ void k_degree(const int* __restrict__ src, const int* __restrict__ dst,
                         int* __restrict__ degO, int* __restrict__ degI, int E) {
  int i = blockIdx.x * blockDim.x + threadIdx.x;
  if (i < E) {
    atomicAdd(&degO[src[i]], 1);
    atomicAdd(&degI[dst[i]], 1);
  }
}

__global__ void k_norm(const int* __restrict__ degO, const int* __restrict__ degI,
                       float* __restrict__ nS, float* __restrict__ nD, int N) {
  int i = blockIdx.x * blockDim.x + threadIdx.x;
  if (i < N) {
    int dO = degO[i]; if (dO < 1) dO = 1;
    int dI = degI[i]; if (dI < 1) dI = 1;
    nS[i] = rsqrtf((float)dO);
    nD[i] = rsqrtf((float)dI);
  }
}

// W [K x Nn] fp32 row-major -> Wt [Nn x K] f16 row-major (transposed, for contiguous B-fragment loads)
__global__ void k_w_to_f16t(const float* __restrict__ w, _Float16* __restrict__ o, int K, int Nn) {
  int idx = blockIdx.x * blockDim.x + threadIdx.x;
  if (idx < K * Nn) {
    int k = idx / Nn;
    int n = idx - k * Nn;
    o[(size_t)n * K + k] = (_Float16)w[idx];
  }
}

// ---------------- fast WMMA GEMM (K compile-time, shapes multiple of 16/64-safe) ----------------
// C[M,Nn] = (A[M,K] @ B[K,Nn]) * rowScale[m];  Bt = B transposed, f16 [Nn][K].
// Block = 128 threads (4 waves); wave computes a 16x16 tile; block covers 16(m) x 64(n).
// Requires: M % 16 == 0, Nn % 16 == 0, K % 32 == 0, K known at compile time.

template <int K>
__global__ void k_gemm_wmma_fast(const float* __restrict__ A, const _Float16* __restrict__ Bt,
                                 float* __restrict__ C, const float* __restrict__ rowScale,
                                 int M, int Nn) {
  const int wave = threadIdx.x >> 5;
  const int lane = threadIdx.x & 31;
  const int m0 = blockIdx.x * 16;
  const int n0 = blockIdx.y * 64 + wave * 16;
  if (n0 >= Nn) return;                    // wave-uniform; EXEC stays all-1s for WMMA

  const int row  = lane & 15;
  const int half = lane >> 4;

  const float*    Arow = A  + (size_t)(m0 + row) * K;   // A row for this lane
  const _Float16* Bcol = Bt + (size_t)(n0 + row) * K;   // B column (contiguous in Bt)

  v8f acc = {};
#pragma unroll
  for (int kb = 0; kb < K; kb += 32) {
    // A 16x32 f16, contiguous k order chosen here: lane loads K = kb+16*half..+15, in element order (matches B).
    const float4 a0 = *(const float4*)(Arow + kb + 16 * half);
    const float4 a1 = *(const float4*)(Arow + kb + 16 * half + 4);
    const float4 a2 = *(const float4*)(Arow + kb + 16 * half + 8);
    const float4 a3 = *(const float4*)(Arow + kb + 16 * half + 12);
    v16h a;
    a[0]  = (_Float16)a0.x; a[1]  = (_Float16)a0.y; a[2]  = (_Float16)a0.z; a[3]  = (_Float16)a0.w;
    a[4]  = (_Float16)a1.x; a[5]  = (_Float16)a1.y; a[6]  = (_Float16)a1.z; a[7]  = (_Float16)a1.w;
    a[8]  = (_Float16)a2.x; a[9]  = (_Float16)a2.y; a[10] = (_Float16)a2.z; a[11] = (_Float16)a2.w;
    a[12] = (_Float16)a3.x; a[13] = (_Float16)a3.y; a[14] = (_Float16)a3.z; a[15] = (_Float16)a3.w;

    // B 32x16 f16, same k order as A: lane loads K = kb+16*half..+15, contiguous in Bt -> one 32B vector load.
    const v16h b = *(const v16h*)(Bcol + kb + 16 * half);

    acc = __builtin_amdgcn_wmma_f32_16x16x32_f16(false, a, false, b, (short)0, acc,
                                                 false, false);
  }

  // rowScale for this lane's 8 output rows: m0+8*half .. +7 (contiguous)
  const float4 s0 = *(const float4*)(rowScale + m0 + 8 * half);
  const float4 s1 = *(const float4*)(rowScale + m0 + 8 * half + 4);
  const float sc[8] = {s0.x, s0.y, s0.z, s0.w, s1.x, s1.y, s1.z, s1.w};

#pragma unroll
  for (int r = 0; r < 8; ++r) {
    const int m = m0 + r + 8 * half;   // C layout: VGPR r -> M=r (lanes 0-15), M=8+r (lanes 16-31)
    C[(size_t)m * Nn + n0 + row] = acc[r] * sc[r];
  }
}

// ---------------- generic WMMA GEMM fallback (runtime K, guarded) ----------------

__global__ void k_gemm_wmma_gen(const float* __restrict__ A, const _Float16* __restrict__ Bt,
                                float* __restrict__ C, const float* __restrict__ rowScale,
                                int M, int K, int Nn) {
  const int wave = threadIdx.x >> 5;
  const int lane = threadIdx.x & 31;
  const int m0 = blockIdx.x * 16;
  const int n0 = blockIdx.y * 64 + wave * 16;
  if (n0 >= Nn) return;

  const int row  = lane & 15;
  const int half = lane >> 4;
  const int mr = (m0 + row < M) ? (m0 + row) : (M - 1);
  const int nc = (n0 + row < Nn) ? (n0 + row) : (Nn - 1);

  v8f acc = {};
  for (int kb = 0; kb < K; kb += 32) {
    v16h a, b;
#pragma unroll
    for (int e = 0; e < 16; ++e) {
      const int ka = 16 * half + e;
      a[e] = (_Float16)A[(size_t)mr * K + kb + ka];
      b[e] = Bt[(size_t)nc * K + kb + 16 * half + e];
    }
    acc = __builtin_amdgcn_wmma_f32_16x16x32_f16(false, a, false, b, (short)0, acc,
                                                 false, false);
  }
#pragma unroll
  for (int r = 0; r < 8; ++r) {
    const int m = m0 + r + 8 * half;
    const int n = n0 + row;
    if (m < M && n < Nn) C[(size_t)m * Nn + n] = acc[r] * rowScale[m];
  }
}

// ---------------- edge scatter: agg[dst[e],f] += hs[src[e],f] * ew[e] (float4) ----------------
// one thread handles 4 features; F power of two, F % 4 == 0.
// shiftE = log2(F/4), so thread idx -> (edge, f4) via shift/mask.

__global__ void k_scatter4(const float* __restrict__ hs, const float* __restrict__ ew,
                           const int* __restrict__ src, const int* __restrict__ dst,
                           float* __restrict__ agg, long long total, int shiftE, int F) {
  long long idx = (long long)blockIdx.x * blockDim.x + threadIdx.x;
  if (idx >= total) return;
  const int e  = (int)(idx >> shiftE);
  const int f4 = (int)(idx & (((long long)1 << shiftE) - 1));
  const float w = ew[e];
  const float4 v = *(const float4*)(hs + (size_t)src[e] * F + f4 * 4);
  float* p = agg + (size_t)dst[e] * F + f4 * 4;
  atomicAdd(p + 0, v.x * w);
  atomicAdd(p + 1, v.y * w);
  atomicAdd(p + 2, v.z * w);
  atomicAdd(p + 3, v.w * w);
}

__global__ void k_scatter_gen(const float* __restrict__ hs, const float* __restrict__ ew,
                              const int* __restrict__ src, const int* __restrict__ dst,
                              float* __restrict__ agg, int E, int F) {
  long long idx = (long long)blockIdx.x * blockDim.x + threadIdx.x;
  if (idx >= (long long)E * F) return;
  int e = (int)(idx / F);
  int f = (int)(idx - (long long)e * F);
  atomicAdd(&agg[(size_t)dst[e] * F + f], hs[(size_t)src[e] * F + f] * ew[e]);
}

// ---------------- post: y = leakyrelu(agg * norm_dst[node]) (float4) ----------------

__global__ void k_post4(const float* __restrict__ agg, const float* __restrict__ nD,
                        float* __restrict__ out, long long total, int shiftN) {
  long long idx = (long long)blockIdx.x * blockDim.x + threadIdx.x;
  if (idx >= total) return;
  const int node = (int)(idx >> shiftN);
  const float s = nD[node];
  float4 v = *(const float4*)(agg + idx * 4);
  v.x = lrelu(v.x * s); v.y = lrelu(v.y * s); v.z = lrelu(v.z * s); v.w = lrelu(v.w * s);
  *(float4*)(out + idx * 4) = v;
}

__global__ void k_post_gen(const float* __restrict__ agg, const float* __restrict__ nD,
                           float* __restrict__ out, int N, int F) {
  long long idx = (long long)blockIdx.x * blockDim.x + threadIdx.x;
  if (idx >= (long long)N * F) return;
  int node = (int)(idx / F);
  out[idx] = lrelu(agg[idx] * nD[node]);
}

// ---------------- pooling ----------------

__global__ void k_pool4(const float* __restrict__ agg2, const float* __restrict__ nD,
                        const int* __restrict__ gid, float* __restrict__ gsum,
                        long long total, int shiftN, int F) {
  long long idx = (long long)blockIdx.x * blockDim.x + threadIdx.x;
  if (idx >= total) return;
  const int node = (int)(idx >> shiftN);
  const int f4   = (int)(idx & (((long long)1 << shiftN) - 1));
  const float s = nD[node];
  float4 v = *(const float4*)(agg2 + idx * 4);
  float* p = gsum + (size_t)gid[node] * F + f4 * 4;
  atomicAdd(p + 0, lrelu(v.x * s));
  atomicAdd(p + 1, lrelu(v.y * s));
  atomicAdd(p + 2, lrelu(v.z * s));
  atomicAdd(p + 3, lrelu(v.w * s));
}

__global__ void k_pool_gen(const float* __restrict__ agg2, const float* __restrict__ nD,
                           const int* __restrict__ gid, float* __restrict__ gsum,
                           int N, int F) {
  long long idx = (long long)blockIdx.x * blockDim.x + threadIdx.x;
  if (idx >= (long long)N * F) return;
  int node = (int)(idx / F);
  int f = (int)(idx - (long long)node * F);
  atomicAdd(&gsum[(size_t)gid[node] * F + f], lrelu(agg2[idx] * nD[node]));
}

__global__ void k_count(const int* __restrict__ gid, float* __restrict__ gcnt, int N) {
  int i = blockIdx.x * blockDim.x + threadIdx.x;
  if (i < N) atomicAdd(&gcnt[gid[i]], 1.0f);
}

__global__ void k_final(const float* __restrict__ gsum, const float* __restrict__ gcnt,
                        float* __restrict__ out, int G, int F) {
  int idx = blockIdx.x * blockDim.x + threadIdx.x;
  if (idx < G * F) out[idx] = gsum[idx] / fmaxf(gcnt[idx / F], 1.0f);
}

// ---------------- host-side launch ----------------

static inline size_t align256(size_t x) { return (x + 255) & ~(size_t)255; }
static inline bool is_pow2(int v) { return v > 0 && (v & (v - 1)) == 0; }
static inline int ilog2(int v) { int s = 0; while ((1 << s) < v) ++s; return s; }

extern "C" void kernel_launch(void* const* d_in, const int* in_sizes, int n_in,
                              void* d_out, int out_size, void* d_ws, size_t ws_size,
                              hipStream_t stream) {
  const float* x   = (const float*)d_in[0];
  const float* ew  = (const float*)d_in[1];
  const float* W1  = (const float*)d_in[2];
  const float* W2  = (const float*)d_in[3];
  const int*   src = (const int*)d_in[4];
  const int*   dst = (const int*)d_in[5];
  const int*   gid = (const int*)d_in[6];

  const int N   = in_sizes[6];            // 50000
  const int E   = in_sizes[1];            // 600000
  const int IN  = in_sizes[0] / N;        // 128
  const int HID = in_sizes[2] / IN;       // 128
  const int RD  = in_sizes[3] / HID;      // 64
  const int G   = out_size / RD;          // 128

  char* base = (char*)d_ws;
  size_t off = 0;
  auto carve = [&](size_t bytes) -> char* {
    char* p = base + off;
    off += align256(bytes);
    return p;
  };
  int*      degO  = (int*)carve((size_t)N * sizeof(int));
  int*      degI  = (int*)carve((size_t)N * sizeof(int));
  float*    normS = (float*)carve((size_t)N * sizeof(float));
  float*    normD = (float*)carve((size_t)N * sizeof(float));
  _Float16* W1t   = (_Float16*)carve((size_t)IN * HID * sizeof(_Float16));  // [HID][IN]
  _Float16* W2t   = (_Float16*)carve((size_t)HID * RD * sizeof(_Float16));  // [RD][HID]
  float*    bufA  = (float*)carve((size_t)N * HID * sizeof(float));  // hs1, then y1
  float*    bufB  = (float*)carve((size_t)N * HID * sizeof(float));  // agg1, then hs2
  float*    bufC  = (float*)carve((size_t)N * RD * sizeof(float));   // agg2
  float*    gsum  = (float*)carve((size_t)G * RD * sizeof(float));
  float*    gcnt  = (float*)carve((size_t)G * sizeof(float));
  (void)ws_size; (void)n_in;

  const int ZB = 4096, T = 256;

  // zero accumulators each call (graph replay must be self-contained)
  k_zero<<<ZB, T, 0, stream>>>((float*)degO, (long long)N);
  k_zero<<<ZB, T, 0, stream>>>((float*)degI, (long long)N);
  k_zero<<<ZB, T, 0, stream>>>(bufB, (long long)N * HID);
  k_zero<<<ZB, T, 0, stream>>>(bufC, (long long)N * RD);
  k_zero<<<ZB, T, 0, stream>>>(gsum, (long long)G * RD);
  k_zero<<<ZB, T, 0, stream>>>(gcnt, (long long)G);

  // degrees + symmetric norms
  k_degree<<<(E + T - 1) / T, T, 0, stream>>>(src, dst, degO, degI, E);
  k_norm<<<(N + T - 1) / T, T, 0, stream>>>(degO, degI, normS, normD, N);

  // weights -> transposed f16
  k_w_to_f16t<<<(IN * HID + T - 1) / T, T, 0, stream>>>(W1, W1t, IN, HID);
  k_w_to_f16t<<<(HID * RD + T - 1) / T, T, 0, stream>>>(W2, W2t, HID, RD);

  const bool fast1 = (IN == 128) && (N % 16 == 0) && (HID % 16 == 0);
  const bool fast2 = (HID == 128) && (N % 16 == 0) && (RD % 16 == 0);

  // ---- layer 1: hs1 = (x @ W1) * norm_src ----
  {
    dim3 grid((N + 15) / 16, (HID + 63) / 64);
    if (fast1)
      k_gemm_wmma_fast<128><<<grid, 128, 0, stream>>>(x, W1t, bufA, normS, N, HID);
    else
      k_gemm_wmma_gen<<<grid, 128, 0, stream>>>(x, W1t, bufA, normS, N, IN, HID);
  }
  // scatter: agg1[dst] += hs1[src] * ew
  if (is_pow2(HID) && HID >= 4) {
    long long tot = (long long)E * (HID / 4);
    k_scatter4<<<(unsigned)((tot + T - 1) / T), T, 0, stream>>>(bufA, ew, src, dst, bufB,
                                                               tot, ilog2(HID / 4), HID);
  } else {
    long long tot = (long long)E * HID;
    k_scatter_gen<<<(unsigned)((tot + T - 1) / T), T, 0, stream>>>(bufA, ew, src, dst, bufB, E, HID);
  }
  // y1 = lrelu(agg1 * norm_dst) -> bufA
  if (is_pow2(HID) && HID >= 4) {
    long long tot = (long long)N * (HID / 4);
    k_post4<<<(unsigned)((tot + T - 1) / T), T, 0, stream>>>(bufB, normD, bufA, tot, ilog2(HID / 4));
  } else {
    long long tot = (long long)N * HID;
    k_post_gen<<<(unsigned)((tot + T - 1) / T), T, 0, stream>>>(bufB, normD, bufA, N, HID);
  }

  // ---- layer 2: hs2 = (y1 @ W2) * norm_src -> bufB ----
  {
    dim3 grid((N + 15) / 16, (RD + 63) / 64);
    if (fast2)
      k_gemm_wmma_fast<128><<<grid, 128, 0, stream>>>(bufA, W2t, bufB, normS, N, RD);
    else
      k_gemm_wmma_gen<<<grid, 128, 0, stream>>>(bufA, W2t, bufB, normS, N, HID, RD);
  }
  // scatter: agg2[dst] += hs2[src] * ew
  if (is_pow2(RD) && RD >= 4) {
    long long tot = (long long)E * (RD / 4);
    k_scatter4<<<(unsigned)((tot + T - 1) / T), T, 0, stream>>>(bufB, ew, src, dst, bufC,
                                                               tot, ilog2(RD / 4), RD);
  } else {
    long long tot = (long long)E * RD;
    k_scatter_gen<<<(unsigned)((tot + T - 1) / T), T, 0, stream>>>(bufB, ew, src, dst, bufC, E, RD);
  }

  // ---- readout: per-graph mean of lrelu(agg2 * norm_dst) ----
  if (is_pow2(RD) && RD >= 4) {
    long long tot = (long long)N * (RD / 4);
    k_pool4<<<(unsigned)((tot + T - 1) / T), T, 0, stream>>>(bufC, normD, gid, gsum,
                                                             tot, ilog2(RD / 4), RD);
  } else {
    long long tot = (long long)N * RD;
    k_pool_gen<<<(unsigned)((tot + T - 1) / T), T, 0, stream>>>(bufC, normD, gid, gsum, N, RD);
  }
  k_count<<<(N + T - 1) / T, T, 0, stream>>>(gid, gcnt, N);
  k_final<<<(G * RD + T - 1) / T, T, 0, stream>>>(gsum, gcnt, (float*)d_out, G, RD);
}
